// Transformer_37598143709213
// MI455X (gfx1250) — matrix operand order repaired and checked by reading, not compiled, not run on hardware
//
#include <hip/hip_runtime.h>
#include <math.h>

// ---------------------------------------------------------------------------
// Types for CDNA5 WMMA (wave32): v_wmma_f32_16x16x32_f16
// ---------------------------------------------------------------------------
typedef __attribute__((ext_vector_type(16))) _Float16 v16h;
typedef __attribute__((ext_vector_type(8)))  _Float16 v8h;
typedef __attribute__((ext_vector_type(8)))  float    v8f;

union Frag16 { v16h v; v8h h[2]; };

__device__ __forceinline__ v8h ld8h(const _Float16* p) { return *(const v8h*)p; }

#define WMMA_F16(a, b, c) \
  __builtin_amdgcn_wmma_f32_16x16x32_f16(false, (a), false, (b), (short)0, (c), false, false)

// ---------------------------------------------------------------------------
// Embedding: x[b,t,:] = tok_emb[idx[b,t]] + pos_emb[t] + class_emb[y[b]]
// ---------------------------------------------------------------------------
__global__ void embed_kernel(const int* __restrict__ idx, const int* __restrict__ y,
                             const float* __restrict__ tok, const float* __restrict__ pos,
                             const float* __restrict__ cls, float* __restrict__ x,
                             int T, int C) {
  int bt = blockIdx.x;
  int b = bt / T, t = bt % T;
  int tk = idx[bt];
  int cl = y[b];
  const float* tr = tok + (size_t)tk * C;
  const float* pr = pos + (size_t)t * C;
  const float* cr = cls + (size_t)cl * C;
  float* xr = x + (size_t)bt * C;
  for (int c = threadIdx.x; c < C; c += blockDim.x)
    xr[c] = tr[c] + pr[c] + cr[c];
}

// ---------------------------------------------------------------------------
// LayerNorm (fp32 in) -> f16 out (GEMM A operand). One row per 256-thr block.
// ---------------------------------------------------------------------------
__global__ void ln_f16_kernel(const float* __restrict__ x, const float* __restrict__ w,
                              const float* __restrict__ b, _Float16* __restrict__ out,
                              int C) {
  __shared__ float sred[8];
  const int row = blockIdx.x, tid = threadIdx.x, lane = tid & 31, wave = tid >> 5;
  const float* xr = x + (size_t)row * C;

  float s = 0.f;
  for (int c = tid; c < C; c += 256) s += xr[c];
  #pragma unroll
  for (int off = 16; off >= 1; off >>= 1) s += __shfl_xor(s, off, 32);
  if (lane == 0) sred[wave] = s;
  __syncthreads();
  float tot = 0.f;
  #pragma unroll
  for (int i = 0; i < 8; ++i) tot += sred[i];
  float mean = tot / (float)C;
  __syncthreads();

  float vs = 0.f;
  for (int c = tid; c < C; c += 256) { float d = xr[c] - mean; vs += d * d; }
  #pragma unroll
  for (int off = 16; off >= 1; off >>= 1) vs += __shfl_xor(vs, off, 32);
  if (lane == 0) sred[wave] = vs;
  __syncthreads();
  float vtot = 0.f;
  #pragma unroll
  for (int i = 0; i < 8; ++i) vtot += sred[i];
  float rstd = rsqrtf(vtot / (float)C + 1e-5f);

  _Float16* orow = out + (size_t)row * C;
  for (int c = tid; c < C; c += 256)
    orow[c] = (_Float16)((xr[c] - mean) * rstd * w[c] + b[c]);
}

// ---------------------------------------------------------------------------
// Weight convert+transpose: W fp32 [K,N] -> WT f16 [N,K] (coalesced writes).
// ---------------------------------------------------------------------------
__global__ void wxpose_kernel(const float* __restrict__ W, _Float16* __restrict__ WT,
                              int K, int N) {
  int i = blockIdx.x * blockDim.x + threadIdx.x;
  if (i >= K * N) return;
  int k = i % K, n = i / K;
  WT[i] = (_Float16)W[(size_t)k * N + n];
}

// ---------------------------------------------------------------------------
// GEMM: out[M,N] = A[M,K](f16) @ WT[N,K](f16)  (+bias, +residual, +GELU)
// block = 256 threads = 8 waves; wave -> 32x64 strip (2 M-tiles x 4 N-tiles);
// block tile 256x64.  REQUIRES K % 64 == 0 (true here: K in {768, 3072}).
// Ping-pong software pipeline (two named fragment sets, no reg-to-reg copies);
// __launch_bounds__(256,2) = 2 waves/SIMD so ~190 VGPRs fit without spilling.
// WMMA layouts per CDNA5 ISA:
//   A 16x32 f16: lane row = lane%16; chunks K = half*8+[0..7], half*8+16+[0..7]
//   B 32x16 f16: lane col = lane%16; chunks K = half*8+[0..7], half*8+16+[0..7] (same as A)
//   C 16x16 f32: vgpr i -> row i + half*8, col = lane%16
// ---------------------------------------------------------------------------
__global__ void __launch_bounds__(256, 2)
gemm_f16_wmma(const _Float16* __restrict__ A,
              const _Float16* __restrict__ WT,
              const float* bias, const float* res,
              float* outF, _Float16* outH,
              int M, int N, int K, int flags /*bit0: gelu*/) {
  const int lane = threadIdx.x & 31;
  const int wave = threadIdx.x >> 5;
  const int half = lane >> 4;
  const int l16  = lane & 15;
  const int m0 = blockIdx.y * 256 + wave * 32;   // 2 row-tiles per wave
  const int n0 = blockIdx.x * 64;

  v8f c[2][4] = {};

  const _Float16* Ap0 = A + (size_t)(m0 + l16) * K + half * 8;
  const _Float16* Ap1 = A + (size_t)(m0 + 16 + l16) * K + half * 8;
  const _Float16* Bp[4];
  #pragma unroll
  for (int t = 0; t < 4; ++t)
    Bp[t] = WT + (size_t)(n0 + t * 16 + l16) * K + half * 8;

  Frag16 xa0, xa1, xb[4];    // set X (even k-steps)
  Frag16 ya0, ya1, yb[4];    // set Y (odd  k-steps)

  // preload k-step 0 into set X
  xa0.h[0] = ld8h(Ap0);  xa0.h[1] = ld8h(Ap0 + 16);
  xa1.h[0] = ld8h(Ap1);  xa1.h[1] = ld8h(Ap1 + 16);
  #pragma unroll
  for (int t = 0; t < 4; ++t) { xb[t].h[0] = ld8h(Bp[t]); xb[t].h[1] = ld8h(Bp[t] + 16); }

  const int nk = K >> 5;                         // even by contract
  for (int ks = 0; ks < nk; ks += 2) {
    {   // prefetch odd step (always exists: nk even)
      const int kn = (ks + 1) * 32;
      ya0.h[0] = ld8h(Ap0 + kn);  ya0.h[1] = ld8h(Ap0 + kn + 16);
      ya1.h[0] = ld8h(Ap1 + kn);  ya1.h[1] = ld8h(Ap1 + kn + 16);
      #pragma unroll
      for (int t = 0; t < 4; ++t) {
        yb[t].h[0] = ld8h(Bp[t] + kn);
        yb[t].h[1] = ld8h(Bp[t] + kn + 16);
      }
    }
    #pragma unroll
    for (int t = 0; t < 4; ++t) {
      c[0][t] = WMMA_F16(xa0.v, xb[t].v, c[0][t]);
      c[1][t] = WMMA_F16(xa1.v, xb[t].v, c[1][t]);
    }
    if (ks + 2 < nk) {   // prefetch next even step into set X
      const int kn = (ks + 2) * 32;
      xa0.h[0] = ld8h(Ap0 + kn);  xa0.h[1] = ld8h(Ap0 + kn + 16);
      xa1.h[0] = ld8h(Ap1 + kn);  xa1.h[1] = ld8h(Ap1 + kn + 16);
      #pragma unroll
      for (int t = 0; t < 4; ++t) {
        xb[t].h[0] = ld8h(Bp[t] + kn);
        xb[t].h[1] = ld8h(Bp[t] + kn + 16);
      }
    }
    #pragma unroll
    for (int t = 0; t < 4; ++t) {
      c[0][t] = WMMA_F16(ya0.v, yb[t].v, c[0][t]);
      c[1][t] = WMMA_F16(ya1.v, yb[t].v, c[1][t]);
    }
  }

  #pragma unroll
  for (int mt = 0; mt < 2; ++mt) {
    #pragma unroll
    for (int t = 0; t < 4; ++t) {
      const int col = n0 + t * 16 + l16;
      const float bv = bias ? bias[col] : 0.f;
      #pragma unroll
      for (int i = 0; i < 8; ++i) {
        const int row = m0 + mt * 16 + i + half * 8;
        float v = c[mt][t][i] + bv;
        if (res) v += res[(size_t)row * N + col];
        if (flags & 1) v = 0.5f * v * (1.f + erff(v * 0.70710678118f));
        if (outF) outF[(size_t)row * N + col] = v;
        if (outH) outH[(size_t)row * N + col] = (_Float16)v;
      }
    }
  }
}

// ---------------------------------------------------------------------------
// Repack qkv f16 [B*T, 3C] -> q [B,H,T,D], k [B,H,T,D], vT [B,H,D,T]
// ---------------------------------------------------------------------------
__global__ void repack_qkv_kernel(const _Float16* __restrict__ qkv,
                                  _Float16* __restrict__ q, _Float16* __restrict__ k,
                                  _Float16* __restrict__ vt,
                                  int B, int T, int H, int D) {
  const int C = H * D;
  int i = blockIdx.x * blockDim.x + threadIdx.x;
  if (i >= B * T * C) return;
  int d = i % D;
  int h = (i / D) % H;
  int t = (i / C) % T;
  int b = i / (T * C);
  size_t src = (size_t)(b * T + t) * (3 * C) + h * D + d;
  size_t bh  = (size_t)(b * H + h);
  q [(bh * T + t) * D + d] = qkv[src];
  k [(bh * T + t) * D + d] = qkv[src + C];
  vt[(bh * D + d) * T + t] = qkv[src + 2 * C];
}

// ---------------------------------------------------------------------------
// Causal flash attention. One wave per 16-query tile per (b,h).
// blockDim = 128 (4 waves), grid = (B*H, T/64).
// S = Q(16xD) @ K^T via WMMA; V B-fragments are loaded right after the S WMMAs
// so the online-softmax VALU work hides their latency; P is staged through LDS
// to become an A-fragment; O += P @ V via WMMA against vT [D,T].
// ---------------------------------------------------------------------------
__global__ void __launch_bounds__(128, 2)
attn_kernel(const _Float16* __restrict__ q,
            const _Float16* __restrict__ kk,
            const _Float16* __restrict__ vt,
            _Float16* __restrict__ out,
            int B, int H, int T, int D, float scale) {
  const int lane = threadIdx.x & 31, wave = threadIdx.x >> 5;
  const int half = lane >> 4, l16 = lane & 15;
  const int bh = blockIdx.x;
  const int b = bh / H, h = bh % H;
  const int qt = blockIdx.y * 4 + wave;         // 16-query tile index
  const int C = H * D;

  const _Float16* Qp = q  + (size_t)bh * T * D;
  const _Float16* Kp = kk + (size_t)bh * T * D;
  const _Float16* Vp = vt + (size_t)bh * D * T;

  __shared__ _Float16 pls[4][16 * 32];
  _Float16* P = pls[wave];

  // Q fragments (rows qt*16 + l16, D = 64 -> two K-steps of 32)
  Frag16 aq0, aq1;
  {
    const _Float16* Qr = Qp + (size_t)(qt * 16 + l16) * D;
    aq0.h[0] = ld8h(Qr + half * 8);       aq0.h[1] = ld8h(Qr + half * 8 + 16);
    aq1.h[0] = ld8h(Qr + 32 + half * 8);  aq1.h[1] = ld8h(Qr + 32 + half * 8 + 16);
  }

  float m[8], lsum[8];
  v8f o[4] = {};
  #pragma unroll
  for (int i = 0; i < 8; ++i) { m[i] = -1e30f; lsum[i] = 0.f; }

  const int nb = (qt + 2) >> 1;                 // key blocks of 32 (causal)
  for (int kb = 0; kb < nb; ++kb) {
    const int kbase = kb * 32;
    v8f s[2] = {};
    #pragma unroll
    for (int st = 0; st < 2; ++st) {
      const _Float16* Kr = Kp + (size_t)(kbase + st * 16 + l16) * D;
      Frag16 b0, b1;
      b0.h[0] = ld8h(Kr + half * 8);      b0.h[1] = ld8h(Kr + half * 8 + 16);
      b1.h[0] = ld8h(Kr + 32 + half * 8); b1.h[1] = ld8h(Kr + 32 + half * 8 + 16);
      s[st] = WMMA_F16(aq0.v, b0.v, s[st]);
      s[st] = WMMA_F16(aq1.v, b1.v, s[st]);
    }

    // Issue V B-fragment loads early: d = t*16+l16 row of vT, keys
    // kbase + half*16 + [0..15].  Softmax below hides their latency.
    Frag16 bv[4];
    #pragma unroll
    for (int t = 0; t < 4; ++t) {
      const _Float16* Vr = Vp + (size_t)(t * 16 + l16) * T + kbase + half * 16;
      bv[t].h[0] = ld8h(Vr);
      bv[t].h[1] = ld8h(Vr + 8);
    }

    // scale + causal mask + row max (rows striped: row = i + half*8)
    float rmax[8];
    #pragma unroll
    for (int i = 0; i < 8; ++i) {
      const int row  = qt * 16 + i + half * 8;
      const int key0 = kbase + l16;
      const int key1 = kbase + 16 + l16;
      float v0 = (key0 <= row) ? s[0][i] * scale : -1e30f;
      float v1 = (key1 <= row) ? s[1][i] * scale : -1e30f;
      s[0][i] = v0; s[1][i] = v1;
      rmax[i] = fmaxf(v0, v1);
    }
    #pragma unroll
    for (int off = 8; off >= 1; off >>= 1)
      #pragma unroll
      for (int i = 0; i < 8; ++i)
        rmax[i] = fmaxf(rmax[i], __shfl_xor(rmax[i], off, 32));

    float rsum[8];
    #pragma unroll
    for (int i = 0; i < 8; ++i) {
      const float mn = fmaxf(m[i], rmax[i]);
      const float alpha = __expf(m[i] - mn);
      const float p0 = __expf(s[0][i] - mn);
      const float p1 = __expf(s[1][i] - mn);
      m[i] = mn;
      s[0][i] = p0; s[1][i] = p1;
      rsum[i] = p0 + p1;
      lsum[i] *= alpha;
      #pragma unroll
      for (int t = 0; t < 4; ++t) o[t][i] *= alpha;
    }
    #pragma unroll
    for (int off = 8; off >= 1; off >>= 1)
      #pragma unroll
      for (int i = 0; i < 8; ++i)
        rsum[i] += __shfl_xor(rsum[i], off, 32);
    #pragma unroll
    for (int i = 0; i < 8; ++i) lsum[i] += rsum[i];

    // stage P (16x32) to LDS: row = i + half*8, col = st*16 + l16
    #pragma unroll
    for (int st = 0; st < 2; ++st)
      #pragma unroll
      for (int i = 0; i < 8; ++i)
        P[(i + half * 8) * 32 + st * 16 + l16] = (_Float16)s[st][i];
    asm volatile("s_wait_dscnt 0" ::: "memory");

    // reload P as an A-fragment: lane row = l16, K = half*16+[0..15]
    Frag16 ap;
    {
      const v8h* Prow = (const v8h*)(P + l16 * 32);
      ap.h[0] = Prow[2 * half];
      ap.h[1] = Prow[2 * half + 1];
    }

    // O += P @ V
    #pragma unroll
    for (int t = 0; t < 4; ++t)
      o[t] = WMMA_F16(ap.v, bv[t].v, o[t]);
  }

  // write normalized O into [B*T, C] f16 (A operand of the proj GEMM)
  #pragma unroll
  for (int i = 0; i < 8; ++i) {
    const int row = qt * 16 + i + half * 8;
    const float inv = 1.f / lsum[i];
    #pragma unroll
    for (int t = 0; t < 4; ++t)
      out[(size_t)(b * T + row) * C + h * D + t * 16 + l16] =
          (_Float16)(o[t][i] * inv);
  }
}

// ---------------------------------------------------------------------------
// Host-side orchestration
// ---------------------------------------------------------------------------
extern "C" void kernel_launch(void* const* d_in, const int* in_sizes, int n_in,
                              void* d_out, int out_size, void* d_ws, size_t ws_size,
                              hipStream_t stream) {
  (void)in_sizes; (void)n_in; (void)out_size; (void)ws_size;
  constexpr int V = 1024, C = 768, H = 12, L = 8, K3 = 3 * C, F = 4 * C;
  constexpr int B = 8, T = 1024, D = C / H;
  constexpr int M = B * T;                 // 8192 tokens

  const int*   idx    = (const int*)  d_in[0];
  const int*   y      = (const int*)  d_in[1];
  const float* tok    = (const float*)d_in[2];
  const float* pos    = (const float*)d_in[3];
  const float* cls    = (const float*)d_in[4];
  const float* ln1_w  = (const float*)d_in[5];
  const float* ln1_b  = (const float*)d_in[6];
  const float* attn_w = (const float*)d_in[7];
  const float* attn_b = (const float*)d_in[8];
  const float* proj_w = (const float*)d_in[9];
  const float* proj_b = (const float*)d_in[10];
  const float* ln2_w  = (const float*)d_in[11];
  const float* ln2_b  = (const float*)d_in[12];
  const float* mlp_w1 = (const float*)d_in[13];
  const float* mlp_b1 = (const float*)d_in[14];
  const float* mlp_w2 = (const float*)d_in[15];
  const float* mlp_b2 = (const float*)d_in[16];
  const float* lnf_w  = (const float*)d_in[17];
  const float* lnf_b  = (const float*)d_in[18];
  const float* head_w = (const float*)d_in[19];

  // workspace carve-up (256B aligned)
  char* ws = (char*)d_ws;
  size_t off = 0;
  auto carve = [&](size_t bytes) -> char* {
    char* p = ws + off;
    off = (off + bytes + 255) & ~(size_t)255;
    return p;
  };
  float*     x_f32  = (float*)    carve((size_t)M * C * 4);
  _Float16*  a_h    = (_Float16*) carve((size_t)M * C * 2);
  _Float16*  qkv_h  = (_Float16*) carve((size_t)M * K3 * 2);
  _Float16*  q_h    = (_Float16*) carve((size_t)M * C * 2);
  _Float16*  k_h    = (_Float16*) carve((size_t)M * C * 2);
  _Float16*  vT_h   = (_Float16*) carve((size_t)M * C * 2);
  _Float16*  attn_h = (_Float16*) carve((size_t)M * C * 2);
  _Float16*  hid_h  = (_Float16*) carve((size_t)M * F * 2);
  _Float16*  wT_h   = (_Float16*) carve((size_t)F * C * 2);   // largest weight

  const float scale = 0.125f;  // 1/sqrt(64)

  auto xpose = [&](const float* W, int K, int N) {
    int n = K * N;
    wxpose_kernel<<<(n + 255) / 256, 256, 0, stream>>>(W, wT_h, K, N);
  };
  auto gemm = [&](const _Float16* A, const float* bias, const float* res,
                  float* outF, _Float16* outH, int N, int K, int flags) {
    dim3 grid(N / 64, M / 256);
    gemm_f16_wmma<<<grid, 256, 0, stream>>>(A, wT_h, bias, res, outF, outH,
                                            M, N, K, flags);
  };

  // embeddings
  embed_kernel<<<M, 256, 0, stream>>>(idx, y, tok, pos, cls, x_f32, T, C);

  for (int l = 0; l < L; ++l) {
    // attention
    ln_f16_kernel<<<M, 256, 0, stream>>>(x_f32, ln1_w + l * C, ln1_b + l * C, a_h, C);
    xpose(attn_w + (size_t)l * C * K3, C, K3);
    gemm(a_h, attn_b + l * K3, nullptr, nullptr, qkv_h, K3, C, 0);
    repack_qkv_kernel<<<(M * C + 255) / 256, 256, 0, stream>>>(
        qkv_h, q_h, k_h, vT_h, B, T, H, D);
    attn_kernel<<<dim3(B * H, T / 64), 128, 0, stream>>>(
        q_h, k_h, vT_h, attn_h, B, H, T, D, scale);
    xpose(proj_w + (size_t)l * C * C, C, C);
    gemm(attn_h, proj_b + l * C, x_f32, x_f32, nullptr, C, C, 0);  // x += attn@W+b

    // MLP
    ln_f16_kernel<<<M, 256, 0, stream>>>(x_f32, ln2_w + l * C, ln2_b + l * C, a_h, C);
    xpose(mlp_w1 + (size_t)l * C * F, C, F);
    gemm(a_h, mlp_b1 + l * F, nullptr, nullptr, hid_h, F, C, 1 /*gelu*/);
    xpose(mlp_w2 + (size_t)l * F * C, F, C);
    gemm(hid_h, mlp_b2 + l * C, x_f32, x_f32, nullptr, C, F, 0);   // x += h@W+b
  }

  // final LN + head -> fp32 logits in d_out
  ln_f16_kernel<<<M, 256, 0, stream>>>(x_f32, lnf_w, lnf_b, a_h, C);
  xpose(head_w, C, V);
  gemm(a_h, nullptr, nullptr, (float*)d_out, nullptr, V, C, 0);
}
